// MultiQueryAttention_81432579932227
// MI455X (gfx1250) — matrix operand order repaired and checked by reading, not compiled, not run on hardware
//
#include <hip/hip_runtime.h>
#include <hip/hip_fp16.h>

typedef __attribute__((ext_vector_type(16))) _Float16 v16h;
typedef __attribute__((ext_vector_type(8)))  float    v8f;

#define BATCH   2
#define SEQ     2048
#define DMODEL  2048
#define NHEADS  16
#define NKV     2
#define HDIM    128
#define KVDIM   (NKV * HDIM)          // 256
#define NREP    (NHEADS / NKV)        // 8
#define SM_SCALE 0.08838834764831845f // 128^-0.5

// ---------------------------------------------------------------------------
// WMMA helper: D = A(16x32 f16) * B(32x16 f16) + C(16x16 f32)
// ---------------------------------------------------------------------------
__device__ __forceinline__ v8f wmma16(v16h a, v16h b, v8f c) {
  return __builtin_amdgcn_wmma_f32_16x16x32_f16(false, a, false, b, (short)0, c,
                                                false, false);
}

// Async 16B global->LDS copy (ASYNCcnt-tracked; LDS written directly, no VGPR
// round-trip). LDS address = low 32 bits of the flat shared pointer (aperture
// rule, ISA 10.2). GV addressing: 64-bit per-lane global address.
__device__ __forceinline__ void async_copy_b128(void* lds, const void* g) {
  const unsigned int ldsoff = (unsigned int)(uintptr_t)lds;
  asm volatile("global_load_async_to_lds_b128 %0, %1, off"
               :: "v"(ldsoff), "v"(g) : "memory");
}
__device__ __forceinline__ void wait_async0() {
  asm volatile("s_wait_asynccnt 0" ::: "memory");
}

// A-fragment (16x32 f16) from LDS row-major [M][K] with pitch (halfs).
// lane m = lane&15 holds row m; elems 0..7 = K=kb..kb+7, 8..15 = K=16+kb..,
// kb = 8*(lane>=16). Two 16B contiguous reads per lane.
__device__ __forceinline__ v16h load_a_lds(const _Float16* base, int pitch) {
  const int lane = threadIdx.x & 31;
  const int m = lane & 15;
  const int kb = (lane >> 4) << 3;
  const _Float16* row = base + m * pitch;
  v16h a;
#pragma unroll
  for (int i = 0; i < 8; ++i) {
    a[i]     = row[kb + i];
    a[8 + i] = row[16 + kb + i];
  }
  return a;
}

// B-fragment (32x16 f16) where LDS holds B^T row-major [N][K] with pitch.
// lane holds column n = lane&15; elems 0..7 = K=kb..kb+7, 8..15 = K=16+kb..,
// kb = 8*(lane>=16). Two 16B contiguous reads per lane.
__device__ __forceinline__ v16h load_bt_lds(const _Float16* base, int pitch) {
  const int lane = threadIdx.x & 31;
  const int n = lane & 15;
  const int kb = (lane >> 4) << 3;
  const _Float16* row = base + n * pitch;
  v16h b;
#pragma unroll
  for (int i = 0; i < 8; ++i) {
    b[i]     = row[kb + i];
    b[8 + i] = row[16 + kb + i];
  }
  return b;
}

// ---------------------------------------------------------------------------
// f32 -> f16 elementwise conversion (vectorized, n multiple of 4)
// ---------------------------------------------------------------------------
__launch_bounds__(256)
__global__ void cvt_kernel(const float* __restrict__ src,
                           _Float16* __restrict__ dst, int n4) {
  const int i = blockIdx.x * blockDim.x + threadIdx.x;
  if (i < n4) {
    const float4 v = ((const float4*)src)[i];
    union { _Float16 h[4]; uint2 u; } p;
    p.h[0] = (_Float16)v.x; p.h[1] = (_Float16)v.y;
    p.h[2] = (_Float16)v.z; p.h[3] = (_Float16)v.w;
    ((uint2*)dst)[i] = p.u;
  }
}

// ---------------------------------------------------------------------------
// Tiled WMMA GEMM, all-f16 operands: C = A @ B
// A: [M,K] f16 row-major (async-staged), B: [K,N] f16 row-major (staged
// transposed in LDS through VGPRs).
// MODE 0: f32 row-major store. MODE 1: f16 row-major. MODE 2: f16 V-transpose
// store -> vT[b][kvh][d][pos].
// Block 256 thr (8 waves), tile 128x128, BK=32, wave tile 32x64 (2x4 WMMA).
// ---------------------------------------------------------------------------
#define BM 128
#define BN 128
#define BK 32
#define AP (BK + 8)  // 40 halfs (80B, 16B-aligned rows)
#define BP (BK + 8)  // Bts pitch

template <int MODE>
__launch_bounds__(256)
__global__ void gemm_f16(const _Float16* __restrict__ A,
                         const _Float16* __restrict__ B,
                         void* __restrict__ Cout, int K, int N) {
  __shared__ _Float16 As[BM][AP];
  __shared__ _Float16 Bts[BN][BP];  // B^T tile: [n][k]

  const int tid  = threadIdx.x;
  const int lane = tid & 31;
  const int wid  = tid >> 5;
  const int wm   = wid & 3;   // 4 waves over M (32 rows each)
  const int wn   = wid >> 2;  // 2 waves over N (64 cols each)
  const int blk_m = blockIdx.y * BM;
  const int blk_n = blockIdx.x * BN;

  v8f acc[2][4] = {};

  for (int k0 = 0; k0 < K; k0 += BK) {
    __syncthreads();
    // Stage A tile 128x32 halfs: 512 16B-chunks, 2 per thread, async to LDS.
#pragma unroll
    for (int c = 0; c < 2; ++c) {
      const int chunk = tid + c * 256;
      const int row = chunk >> 2;
      const int col = (chunk & 3) * 8;
      const _Float16* gp = &A[(size_t)(blk_m + row) * K + k0 + col];
      async_copy_b128(&As[row][col], gp);
      if (k0 + BK < K) __builtin_prefetch(gp + BK, 0, 3);
    }
    // Stage B tile 32x128 halfs transposed into Bts[n][k]: 512 chunks.
#pragma unroll
    for (int c = 0; c < 2; ++c) {
      const int chunk = tid + c * 256;
      const int row = chunk >> 4;          // k within tile
      const int nc  = (chunk & 15) * 8;    // n within tile
      const _Float16* gp = &B[(size_t)(k0 + row) * N + blk_n + nc];
      const uint4 u = *(const uint4*)gp;
      if (k0 + BK < K) __builtin_prefetch(gp + (size_t)BK * N, 0, 3);
      union { uint4 u; _Float16 h[8]; } p;
      p.u = u;
#pragma unroll
      for (int i = 0; i < 8; ++i) Bts[nc + i][row] = p.h[i];
    }
    wait_async0();
    __syncthreads();

    v16h a0 = load_a_lds(&As[wm * 32][0], AP);
    v16h a1 = load_a_lds(&As[wm * 32 + 16][0], AP);
#pragma unroll
    for (int j = 0; j < 4; ++j) {
      v16h bj = load_bt_lds(&Bts[wn * 64 + j * 16][0], BP);
      acc[0][j] = wmma16(a0, bj, acc[0][j]);
      acc[1][j] = wmma16(a1, bj, acc[1][j]);
    }
  }

  // C/D layout: lane n = lane&15; VGPR r -> row mbase+r, mbase = 8*(lane>=16)
  const int cn    = lane & 15;
  const int mbase = (lane >> 4) << 3;
#pragma unroll
  for (int i = 0; i < 2; ++i)
#pragma unroll
    for (int j = 0; j < 4; ++j) {
      const int m0 = blk_m + wm * 32 + i * 16 + mbase;
      const int n0 = blk_n + wn * 64 + j * 16 + cn;
#pragma unroll
      for (int r = 0; r < 8; ++r) {
        const float val = acc[i][j][r];
        const int m = m0 + r;
        if (MODE == 0) {
          ((float*)Cout)[(size_t)m * N + n0] = val;
        } else if (MODE == 1) {
          ((_Float16*)Cout)[(size_t)m * N + n0] = (_Float16)val;
        } else {
          const int b = m >> 11, pos = m & (SEQ - 1);
          const int kvh = n0 >> 7, d = n0 & (HDIM - 1);
          ((_Float16*)Cout)[(size_t)((b * NKV + kvh) * HDIM + d) * SEQ + pos] =
              (_Float16)val;
        }
      }
    }
}

// ---------------------------------------------------------------------------
// Flash attention (online softmax), all-f16 inputs. Block = 128 thr (4 waves),
// each wave owns 16 queries of head (b,h); 32-key chunks of K/V shared in LDS
// via async copies.
// Q: [b][n][h*128+d] f16, Kc: [b][n][kvh*128+d] f16, VT: [b][kvh][d][n] f16.
// Out: [b][n][h*128+d] f16.
// ---------------------------------------------------------------------------
#define KP (HDIM + 8)  // 136 halfs
#define VP (32 + 8)    // 40
#define PP (32 + 8)    // 40

__launch_bounds__(128)
__global__ void attn_kernel(const _Float16* __restrict__ Q,
                            const _Float16* __restrict__ Kc,
                            const _Float16* __restrict__ VT,
                            _Float16* __restrict__ O) {
  __shared__ _Float16 Ks[32][KP];     // [key][d]
  __shared__ _Float16 Vs[HDIM][VP];   // [d][key]
  __shared__ _Float16 Ps[4][16][PP];  // per-wave P patch

  const int tid  = threadIdx.x;
  const int lane = tid & 31;
  const int wid  = tid >> 5;
  const int b    = blockIdx.z;
  const int h    = blockIdx.y;
  const int kvh  = h / NREP;
  const int q0   = blockIdx.x * 64 + wid * 16;

  const _Float16* qbase  = Q + (size_t)(b * SEQ + q0) * DMODEL + h * HDIM;
  const _Float16* kbase  = Kc + (size_t)(b * SEQ) * KVDIM + kvh * HDIM;
  const _Float16* vtbase = VT + (size_t)((b * NKV + kvh) * HDIM) * SEQ;

  // Q fragments: 16x128 as 4 A-fragments (K-chunks of 32)
  v16h qf[4];
  {
    const int m = lane & 15;
    const int kb = (lane >> 4) << 3;
    const _Float16* row = qbase + (size_t)m * DMODEL;
#pragma unroll
    for (int kk = 0; kk < 4; ++kk)
#pragma unroll
      for (int i = 0; i < 8; ++i) {
        qf[kk][i]     = row[kk * 32 + kb + i];
        qf[kk][8 + i] = row[kk * 32 + 16 + kb + i];
      }
  }

  v8f o[8] = {};
  float mrun[8], lrun[8];
#pragma unroll
  for (int r = 0; r < 8; ++r) { mrun[r] = -1e30f; lrun[r] = 0.f; }

  const int n     = lane & 15;
  const int mbase = (lane >> 4) << 3;

  for (int j0 = 0; j0 < SEQ; j0 += 32) {
    __syncthreads();
    // K chunk: 32 keys x 128 halfs = 512 16B-chunks, 4 per thread (async).
#pragma unroll
    for (int c = 0; c < 4; ++c) {
      const int chunk = tid + c * 128;
      const int row = chunk >> 4;
      const int col = (chunk & 15) * 8;
      const _Float16* gp = &kbase[(size_t)(j0 + row) * KVDIM + col];
      async_copy_b128(&Ks[row][col], gp);
      if (j0 + 32 < SEQ) __builtin_prefetch(gp + (size_t)32 * KVDIM, 0, 3);
    }
    // V^T chunk: 128 d x 32 halfs = 512 16B-chunks, 4 per thread (async).
#pragma unroll
    for (int c = 0; c < 4; ++c) {
      const int chunk = tid + c * 128;
      const int row = chunk >> 2;
      const int col = (chunk & 3) * 8;
      const _Float16* gp = &vtbase[(size_t)row * SEQ + j0 + col];
      async_copy_b128(&Vs[row][col], gp);
      if (j0 + 32 < SEQ) __builtin_prefetch(gp + 32, 0, 3);
    }
    wait_async0();
    __syncthreads();

    // S = Q @ K^T : two 16x16 tiles over d-chunks of 32
    v8f s0 = {}, s1 = {};
#pragma unroll
    for (int kk = 0; kk < 4; ++kk) {
      v16h bk0 = load_bt_lds(&Ks[0][kk * 32], KP);
      v16h bk1 = load_bt_lds(&Ks[16][kk * 32], KP);
      s0 = wmma16(qf[kk], bk0, s0);
      s1 = wmma16(qf[kk], bk1, s1);
    }

    float p0[8], p1[8], alpha[8], mc[8];
#pragma unroll
    for (int r = 0; r < 8; ++r) {
      p0[r] = s0[r] * SM_SCALE;
      p1[r] = s1[r] * SM_SCALE;
      mc[r] = fmaxf(p0[r], p1[r]);
    }
    // row max across the 16 lanes of each half-wave (rows live per half)
#pragma unroll
    for (int off = 1; off < 16; off <<= 1)
#pragma unroll
      for (int r = 0; r < 8; ++r)
        mc[r] = fmaxf(mc[r], __shfl_xor(mc[r], off, 32));

#pragma unroll
    for (int r = 0; r < 8; ++r) {
      const float mn = fmaxf(mrun[r], mc[r]);
      alpha[r] = __expf(mrun[r] - mn);
      mrun[r] = mn;
      p0[r] = __expf(p0[r] - mn);
      p1[r] = __expf(p1[r] - mn);
    }
    float rs[8];
#pragma unroll
    for (int r = 0; r < 8; ++r) rs[r] = p0[r] + p1[r];
#pragma unroll
    for (int off = 1; off < 16; off <<= 1)
#pragma unroll
      for (int r = 0; r < 8; ++r) rs[r] += __shfl_xor(rs[r], off, 32);
#pragma unroll
    for (int r = 0; r < 8; ++r) lrun[r] = lrun[r] * alpha[r] + rs[r];
#pragma unroll
    for (int nt = 0; nt < 8; ++nt)
#pragma unroll
      for (int r = 0; r < 8; ++r) o[nt][r] = o[nt][r] * alpha[r];

    // P: C-layout -> per-wave LDS patch -> A-fragment layout
#pragma unroll
    for (int r = 0; r < 8; ++r) {
      Ps[wid][mbase + r][n]      = (_Float16)p0[r];
      Ps[wid][mbase + r][16 + n] = (_Float16)p1[r];
    }
    asm volatile("" ::: "memory");  // keep DS store->load order (HW in-order)
    v16h pf = load_a_lds(&Ps[wid][0][0], PP);

    // O += P(16x32) @ V(32x128)
#pragma unroll
    for (int nt = 0; nt < 8; ++nt) {
      v16h vb = load_bt_lds(&Vs[nt * 16][0], VP);
      o[nt] = wmma16(pf, vb, o[nt]);
    }
  }

#pragma unroll
  for (int r = 0; r < 8; ++r) lrun[r] = 1.f / lrun[r];
#pragma unroll
  for (int nt = 0; nt < 8; ++nt)
#pragma unroll
    for (int r = 0; r < 8; ++r)
      O[(size_t)(b * SEQ + q0 + mbase + r) * DMODEL + h * HDIM + nt * 16 + n] =
          (_Float16)(o[nt][r] * lrun[r]);
}

// ---------------------------------------------------------------------------
extern "C" void kernel_launch(void* const* d_in, const int* in_sizes, int n_in,
                              void* d_out, int out_size, void* d_ws,
                              size_t ws_size, hipStream_t stream) {
  (void)in_sizes; (void)n_in; (void)out_size; (void)ws_size;
  const float* x  = (const float*)d_in[0];
  const float* Wq = (const float*)d_in[1];
  const float* Wk = (const float*)d_in[2];
  const float* Wv = (const float*)d_in[3];
  const float* Wo = (const float*)d_in[4];
  float* out = (float*)d_out;

  // Workspace carve-out (all f16). Sizes in halfs.
  const size_t SZ_X  = (size_t)BATCH * SEQ * DMODEL;  // 8.4M
  const size_t SZ_WQ = (size_t)DMODEL * DMODEL;       // 4.2M
  const size_t SZ_WK = (size_t)DMODEL * KVDIM;        // 0.5M
  _Float16* xh    = (_Float16*)d_ws;
  _Float16* wqh   = xh + SZ_X;
  _Float16* wkh   = wqh + SZ_WQ;
  _Float16* wvh   = wkh + SZ_WK;
  _Float16* woh   = wvh + SZ_WK;
  _Float16* qh    = woh + SZ_WQ;
  _Float16* kh    = qh + SZ_X;
  _Float16* vTh   = kh + (size_t)BATCH * SEQ * KVDIM;
  _Float16* attnh = vTh + (size_t)BATCH * NKV * HDIM * SEQ;

  // 1) One-time f32 -> f16 conversion of activations + weights.
  cvt_kernel<<<dim3((int)(SZ_X / 4 / 256)), dim3(256), 0, stream>>>(x, xh, (int)(SZ_X / 4));
  cvt_kernel<<<dim3((int)(SZ_WQ / 4 / 256)), dim3(256), 0, stream>>>(Wq, wqh, (int)(SZ_WQ / 4));
  cvt_kernel<<<dim3((int)(SZ_WK / 4 / 256)), dim3(256), 0, stream>>>(Wk, wkh, (int)(SZ_WK / 4));
  cvt_kernel<<<dim3((int)(SZ_WK / 4 / 256)), dim3(256), 0, stream>>>(Wv, wvh, (int)(SZ_WK / 4));
  cvt_kernel<<<dim3((int)(SZ_WQ / 4 / 256)), dim3(256), 0, stream>>>(Wo, woh, (int)(SZ_WQ / 4));

  const dim3 gblk(256);
  const int MT = BATCH * SEQ / BM;  // 32 M-tiles

  // 2) Projections (f16 in, f16 out; V written transposed).
  gemm_f16<1><<<dim3(DMODEL / BN, MT), gblk, 0, stream>>>(xh, wqh, qh, DMODEL, DMODEL);
  gemm_f16<1><<<dim3(KVDIM / BN, MT), gblk, 0, stream>>>(xh, wkh, kh, DMODEL, KVDIM);
  gemm_f16<2><<<dim3(KVDIM / BN, MT), gblk, 0, stream>>>(xh, wvh, vTh, DMODEL, KVDIM);

  // 3) Flash attention.
  attn_kernel<<<dim3(SEQ / 64, NHEADS, BATCH), dim3(128), 0, stream>>>(qh, kh, vTh, attnh);

  // 4) Output projection (f16 in, f32 out).
  gemm_f16<0><<<dim3(DMODEL / BN, MT), gblk, 0, stream>>>(attnh, woh, out, DMODEL, DMODEL);
}
